// RotarySelfAttention_1494648619014
// MI455X (gfx1250) — matrix operand order repaired and checked by reading, not compiled, not run on hardware
//
#include <hip/hip_runtime.h>
#include <hip/hip_bf16.h>
#include <math.h>

// Problem constants (match reference)
#define BATCH  4
#define SEQ    2048
#define DMODEL 1024
#define NHEAD  16
#define DHEAD  64
#define DINNER 1024   // NHEAD*DHEAD
#define LN_EPS 1e-5f

typedef __attribute__((ext_vector_type(16))) _Float16 v16h;
typedef __attribute__((ext_vector_type(8)))  _Float16 v8h;
typedef __attribute__((ext_vector_type(8)))  float    v8f;
typedef __attribute__((ext_vector_type(4)))  unsigned int v4u;
typedef __attribute__((ext_vector_type(4)))  int      v4i;
typedef __attribute__((ext_vector_type(8)))  int      v8i;

#define CAT16(lo, hi) __builtin_shufflevector(lo, hi, 0,1,2,3,4,5,6,7,8,9,10,11,12,13,14,15)

#if __has_builtin(__builtin_amdgcn_tensor_load_to_lds)
#define HAVE_TDM 1
#else
#define HAVE_TDM 0
#endif

__device__ __forceinline__ void wait_tensorcnt0() {
#if __has_builtin(__builtin_amdgcn_s_wait_tensorcnt)
    __builtin_amdgcn_s_wait_tensorcnt(0);
#else
    asm volatile("s_wait_tensorcnt 0x0" ::: "memory");
#endif
}

// ---------------------------------------------------------------------------
// 0) f32 -> f16 elementwise conversion (for weights)
// ---------------------------------------------------------------------------
__global__ void cvt_f32_f16_kernel(const float* __restrict__ src,
                                   _Float16* __restrict__ dst, int n) {
    int i = blockIdx.x * 256 + threadIdx.x;
    if (i < n) dst[i] = (_Float16)src[i];
}

// ---------------------------------------------------------------------------
// 1) LayerNorm over DMODEL, output f16. One block (256 thr) per row.
// ---------------------------------------------------------------------------
__global__ void ln_kernel(const float* __restrict__ x,
                          const float* __restrict__ gamma,
                          const float* __restrict__ beta,
                          _Float16* __restrict__ xn) {
    __shared__ float red[256];
    const int row = blockIdx.x, t = threadIdx.x;
    const float* xr = x + (size_t)row * DMODEL;
    float v[4];
    float s = 0.0f;
    for (int i = 0; i < 4; ++i) { v[i] = xr[t + i * 256]; s += v[i]; }
    red[t] = s; __syncthreads();
    for (int off = 128; off > 0; off >>= 1) {
        if (t < off) red[t] += red[t + off];
        __syncthreads();
    }
    const float mu = red[0] * (1.0f / DMODEL);
    __syncthreads();
    s = 0.0f;
    for (int i = 0; i < 4; ++i) { float d = v[i] - mu; s += d * d; }
    red[t] = s; __syncthreads();
    for (int off = 128; off > 0; off >>= 1) {
        if (t < off) red[t] += red[t + off];
        __syncthreads();
    }
    const float rs = rsqrtf(red[0] * (1.0f / DMODEL) + LN_EPS);
    for (int i = 0; i < 4; ++i) {
        int c = t + i * 256;
        xn[(size_t)row * DMODEL + c] = (_Float16)((v[i] - mu) * rs * gamma[c] + beta[c]);
    }
}

// ---------------------------------------------------------------------------
// 2) Tiled f16 WMMA GEMM:  C[M,Nc] = A[M,K] @ B[K,Nc]
//    Workgroup tile 128x128, 8 waves (4 over M x 2 over N), each wave 32x64.
//    Double-buffered LDS; A tile staged by the Tensor Data Mover
//    (tensor_load_to_lds, TENSORcnt), B tile by coalesced loads + LDS
//    transpose. Output either f16 (Ch) or f32+bias (Cf).
// ---------------------------------------------------------------------------
__global__ void gemm_f16_kernel(const _Float16* __restrict__ A,
                                const _Float16* __restrict__ Bm,
                                _Float16* __restrict__ Ch,
                                float* __restrict__ Cf,
                                const float* __restrict__ bias,
                                int M, int K, int Nc) {
    __shared__ __attribute__((aligned(16))) _Float16 As[2][128][32];   // [buf][m][k]
    __shared__ __attribute__((aligned(16))) _Float16 Bs[2][128][32];   // [buf][n][k]
    const int t = threadIdx.x;
    const int lane = t & 31, w = t >> 5;
    const int m16 = lane & 15, g = lane >> 4;
    const int mw = w >> 1, nw = w & 1;
    const int mBase = blockIdx.y * 128, nBase = blockIdx.x * 128;
    const int nkb = K >> 5;

    // ---- A tile fill: TDM descriptor load (wave-level) or sync fallback ----
    auto fillA = [&](int kb, int buf) {
#if HAVE_TDM
        if (t == 0) {
            // D# group0: count=1 | lds_addr | global_addr(57b) | type=2
            unsigned lds_addr = (unsigned)(unsigned long long)(void*)&As[buf][0][0]; // low 32b of a
                                                                                    // generic LDS ptr
                                                                                    // = LDS byte offset
            unsigned long long ga = (unsigned long long)A
                                  + ((size_t)mBase * K + (size_t)kb * 32) * 2;
            v4u g0 = { 1u, lds_addr, (unsigned)ga,
                       (unsigned)((ga >> 32) & 0x01FFFFFFu) | 0x80000000u /*type=2*/ };
            // D# group1: data_size=2B | tensor_dim0=K | tensor_dim1=M |
            //            tile_dim0=32 | tile_dim1=128 | dim0_stride=K
            v8i g1 = { (int)(1u << 16),
                       (int)((unsigned)(K & 0xFFFF) << 16),
                       (int)(((unsigned)(K >> 16) & 0xFFFFu) | ((unsigned)(M & 0xFFFF) << 16)),
                       (int)(((unsigned)(M >> 16) & 0xFFFFu) | (32u << 16)),
                       (int)128,
                       (int)K,
                       0, 0 };
            v4i z4 = {};
#if defined(__clang_major__) && (__clang_major__ >= 23)
            v8i z8 = {};
            __builtin_amdgcn_tensor_load_to_lds(g0, g1, z4, z4, z8, 0);
#else
            __builtin_amdgcn_tensor_load_to_lds(g0, g1, z4, z4, 0);
#endif
        }
#else
        for (int c = 0; c < 2; ++c) {
            int lin8 = t + c * 256;                 // unit = 8 f16
            int row = lin8 >> 2, k8 = lin8 & 3;
            const _Float16* src = A + (size_t)(mBase + row) * K + kb * 32 + k8 * 8;
            *(uint4*)&As[buf][row][k8 * 8] = *(const uint4*)src;
        }
#endif
    };

    // ---- B tile fill: coalesced global b128 loads, transpose-scatter to LDS
    auto fillB = [&](int kb, int buf) {
        for (int c = 0; c < 2; ++c) {
            int lin8 = t + c * 256;
            int k = lin8 >> 4, n8 = lin8 & 15;
            const _Float16* src = Bm + (size_t)(kb * 32 + k) * Nc + nBase + n8 * 8;
            v8h bv = *(const v8h*)src;
            for (int i = 0; i < 8; ++i) Bs[buf][n8 * 8 + i][k] = bv[i];
            if (kb + 1 < nkb) __builtin_prefetch(src + (size_t)32 * Nc, 0, 1);
        }
    };

    v8f acc[2][4] = {};

    fillA(0, 0);
    fillB(0, 0);
#if HAVE_TDM
    if (t == 0) wait_tensorcnt0();
#endif
    __syncthreads();

    int cb = 0;
    for (int kb = 0; kb < nkb; ++kb) {
        const int nb = cb ^ 1;
        if (kb + 1 < nkb) {     // issue next tile while computing this one
            fillA(kb + 1, nb);
            fillB(kb + 1, nb);
        }

        v16h af[2], bf[4];
        for (int mt = 0; mt < 2; ++mt) {            // A frag: elems 0-7 K=g*8+i, 8-15 K=16+g*8+i
            int row = mw * 32 + mt * 16 + m16;
            v8h lo = *(const v8h*)&As[cb][row][g * 8];
            v8h hi = *(const v8h*)&As[cb][row][16 + g * 8];
            af[mt] = CAT16(lo, hi);
        }
        for (int nt = 0; nt < 4; ++nt) {            // B frag: elems 0-7 K=g*8+i, 8-15 K=16+g*8+i
            int col = nw * 64 + nt * 16 + m16;
            v8h lo = *(const v8h*)&Bs[cb][col][g * 8];
            v8h hi = *(const v8h*)&Bs[cb][col][16 + g * 8];
            bf[nt] = CAT16(lo, hi);
        }
        for (int mt = 0; mt < 2; ++mt)
            for (int nt = 0; nt < 4; ++nt)
                acc[mt][nt] = __builtin_amdgcn_wmma_f32_16x16x32_f16(
                    false, af[mt], false, bf[nt], (short)0, acc[mt][nt], false, false);

#if HAVE_TDM
        if (t == 0 && kb + 1 < nkb) wait_tensorcnt0();
#endif
        __syncthreads();
        cb = nb;
    }

    for (int mt = 0; mt < 2; ++mt)
        for (int nt = 0; nt < 4; ++nt) {
            int gm0 = mBase + mw * 32 + mt * 16 + 8 * g;   // C layout: elem r -> row 8g+r
            int gn  = nBase + nw * 64 + nt * 16 + m16;     //            lane col = L%16
            for (int r = 0; r < 8; ++r) {
                size_t idx = (size_t)(gm0 + r) * Nc + gn;
                if (Ch) Ch[idx] = (_Float16)acc[mt][nt][r];
                else    Cf[idx] = acc[mt][nt][r] + (bias ? bias[gn] : 0.0f);
            }
        }
}

// ---------------------------------------------------------------------------
// 3) Rotary on q,k,v. qkv f16 [B,N,3*DINNER] -> q,k [B,H,N,DH], v^T [B,H,DH,N]
// ---------------------------------------------------------------------------
__global__ void rotary_kernel(const _Float16* __restrict__ qkv,
                              const float* __restrict__ fr,
                              _Float16* __restrict__ qo,
                              _Float16* __restrict__ ko,
                              _Float16* __restrict__ vt) {
    int tid = blockIdx.x * 256 + threadIdx.x;
    int i2 = tid & 31;
    int h  = (tid >> 5) & 15;
    int n  = (tid >> 9) & 2047;
    int b  = tid >> 20;
    int d  = 2 * i2;
    size_t fb = ((size_t)b * SEQ + n) * DHEAD + d;
    float f0 = fr[fb], f1 = fr[fb + 1];
    float c0 = cosf(f0), s0 = sinf(f0), c1 = cosf(f1), s1 = sinf(f1);
    size_t src  = ((size_t)b * SEQ + n) * (3 * DINNER) + h * DHEAD + d;
    size_t dstq = (((size_t)b * NHEAD + h) * SEQ + n) * DHEAD + d;
    {   // q
        float x0 = (float)qkv[src], x1 = (float)qkv[src + 1];
        qo[dstq]     = (_Float16)(x0 * c0 - x1 * s0);
        qo[dstq + 1] = (_Float16)(x1 * c1 + x0 * s1);
    }
    {   // k
        float x0 = (float)qkv[src + DINNER], x1 = (float)qkv[src + DINNER + 1];
        ko[dstq]     = (_Float16)(x0 * c0 - x1 * s0);
        ko[dstq + 1] = (_Float16)(x1 * c1 + x0 * s1);
    }
    {   // v, stored transposed for contiguous P@V B-fragments
        float x0 = (float)qkv[src + 2 * DINNER], x1 = (float)qkv[src + 2 * DINNER + 1];
        size_t vb = ((size_t)b * NHEAD + h) * DHEAD;
        vt[(vb + d) * SEQ + n]     = (_Float16)(x0 * c0 - x1 * s0);
        vt[(vb + d + 1) * SEQ + n] = (_Float16)(x1 * c1 + x0 * s1);
    }
}

// ---------------------------------------------------------------------------
// 4) Flash attention. One wave per 16 query rows, KV streamed 32 at a time.
//    Block = 128 threads (4 waves). Grid = B*H*(SEQ/64).
// ---------------------------------------------------------------------------
__global__ void attn_kernel(const _Float16* __restrict__ Q,   // [BH][N][DH]
                            const _Float16* __restrict__ Kh_, // [BH][N][DH]
                            const _Float16* __restrict__ Vt_, // [BH][DH][N]
                            const unsigned char* __restrict__ mask, // [B][N] bool
                            float* __restrict__ O) {          // [B][N][H][DH]
    __shared__ __attribute__((aligned(16))) _Float16 Plds[4][16][32];
    const int lane = threadIdx.x & 31;
    const int w = threadIdx.x >> 5;
    const int m16 = lane & 15;
    const int g = lane >> 4;
    const int bh = blockIdx.x >> 5;          // SEQ/64 = 32 q-chunks per head
    const int qc = blockIdx.x & 31;
    const int b = bh >> 4, h = bh & 15;
    const int q0 = qc * 64 + w * 16;

    const _Float16* Qh = Q   + (size_t)bh * SEQ * DHEAD;
    const _Float16* Kh = Kh_ + (size_t)bh * SEQ * DHEAD;
    const _Float16* Vh = Vt_ + (size_t)bh * DHEAD * SEQ;
    const unsigned char* mb = mask + (size_t)b * SEQ;

    // Q fragments (A layout), reused for every KV tile
    v16h aQ[2];
    for (int kc = 0; kc < 2; ++kc) {
        const _Float16* p = Qh + (size_t)(q0 + m16) * DHEAD + kc * 32 + g * 8;
        v8h lo = *(const v8h*)p;
        v8h hi = *(const v8h*)(p + 16);
        aQ[kc] = CAT16(lo, hi);
    }

    v8f accO[4] = {};                 // 16x64 f32 output accumulator
    float mrow[8], lrow[8];
    for (int r = 0; r < 8; ++r) { mrow[r] = -3.0e38f; lrow[r] = 0.0f; }
    const float scale = 0.125f;       // DH^-0.5

    for (int kv = 0; kv < SEQ; kv += 32) {
        v8f S[2];
        float mbias[2];
        for (int nh = 0; nh < 2; ++nh) {
            v8f acc = {};
            for (int kc = 0; kc < 2; ++kc) {
                // B frag of K^T: col = kv index, elems 0-7 -> K-dim g*8+i, 8-15 -> 16+g*8+i
                const _Float16* p = Kh + (size_t)(kv + nh * 16 + m16) * DHEAD + kc * 32 + g * 8;
                v8h lo = *(const v8h*)p;
                v8h hi = *(const v8h*)(p + 16);
                v16h bK = CAT16(lo, hi);
                acc = __builtin_amdgcn_wmma_f32_16x16x32_f16(
                    false, aQ[kc], false, bK, (short)0, acc, false, false);
            }
            S[nh] = acc;
            mbias[nh] = mb[kv + nh * 16 + m16] ? -1.0e30f : 0.0f;
        }
        // online softmax: a score row lives across 16 lanes -> width-16 shuffles
        for (int r = 0; r < 8; ++r) {
            float v0 = S[0][r] * scale + mbias[0];
            float v1 = S[1][r] * scale + mbias[1];
            float mx = fmaxf(v0, v1);
            for (int off = 8; off >= 1; off >>= 1)
                mx = fmaxf(mx, __shfl_xor(mx, off, 16));
            float mnew = fmaxf(mrow[r], mx);
            float corr = __expf(mrow[r] - mnew);
            float p0 = __expf(v0 - mnew);
            float p1 = __expf(v1 - mnew);
            float ps = p0 + p1;
            for (int off = 8; off >= 1; off >>= 1)
                ps += __shfl_xor(ps, off, 16);
            lrow[r] = lrow[r] * corr + ps;
            mrow[r] = mnew;
            for (int nt = 0; nt < 4; ++nt) accO[nt][r] *= corr;
            int m = 8 * g + r;                       // C layout row
            Plds[w][m][m16]      = (_Float16)p0;
            Plds[w][m][16 + m16] = (_Float16)p1;
        }
        // reshape P: C layout -> A layout via per-wave LDS tile (same-wave DS order)
        v8h plo = *(const v8h*)&Plds[w][m16][g * 8];
        v8h phi = *(const v8h*)&Plds[w][m16][16 + g * 8];
        v16h aP = CAT16(plo, phi);
        for (int nt = 0; nt < 4; ++nt) {
            // B frag of V: col = head dim, elems 0-7 -> kv g*8+i, 8-15 -> kv 16+g*8+i
            const _Float16* p = Vh + (size_t)(nt * 16 + m16) * SEQ + kv + g * 8;
            v8h lo = *(const v8h*)p;
            v8h hi = *(const v8h*)(p + 16);
            v16h bV = CAT16(lo, hi);
            accO[nt] = __builtin_amdgcn_wmma_f32_16x16x32_f16(
                false, aP, false, bV, (short)0, accO[nt], false, false);
        }
    }
    // normalize and store [B,N,H,DH] fp32
    for (int r = 0; r < 8; ++r) {
        float inv = 1.0f / lrow[r];
        int qi = q0 + 8 * g + r;
        for (int nt = 0; nt < 4; ++nt) {
            int d = nt * 16 + m16;
            O[(((size_t)(b * SEQ + qi)) * NHEAD + h) * DHEAD + d] = accO[nt][r] * inv;
        }
    }
}

// ---------------------------------------------------------------------------
// 5) Inverse rotary (apply_rotary with -freqs), pack to f16 [B,N,DINNER]
// ---------------------------------------------------------------------------
__global__ void unrotary_kernel(const float* __restrict__ o,   // [B,N,H,DH] f32
                                const float* __restrict__ fr,
                                _Float16* __restrict__ ob) {   // [B,N,DINNER] f16
    int tid = blockIdx.x * 256 + threadIdx.x;
    int i2 = tid & 31;
    int h  = (tid >> 5) & 15;
    int n  = (tid >> 9) & 2047;
    int b  = tid >> 20;
    int d  = 2 * i2;
    size_t fb = ((size_t)b * SEQ + n) * DHEAD + d;
    float f0 = fr[fb], f1 = fr[fb + 1];
    float c0 = cosf(f0), s0 = sinf(f0), c1 = cosf(f1), s1 = sinf(f1);
    size_t src = (((size_t)b * SEQ + n) * NHEAD + h) * DHEAD + d;
    float x0 = o[src], x1 = o[src + 1];
    size_t dst = ((size_t)b * SEQ + n) * DINNER + h * DHEAD + d;
    ob[dst]     = (_Float16)(x0 * c0 + x1 * s0);   // cos(-f)=cos f, sin(-f)=-sin f
    ob[dst + 1] = (_Float16)(x1 * c1 - x0 * s1);
}

// ---------------------------------------------------------------------------
// launch
// ---------------------------------------------------------------------------
extern "C" void kernel_launch(void* const* d_in, const int* in_sizes, int n_in,
                              void* d_out, int out_size, void* d_ws, size_t ws_size,
                              hipStream_t stream) {
    const float*         x     = (const float*)d_in[0];
    const float*         freqs = (const float*)d_in[1];
    const unsigned char* mask  = (const unsigned char*)d_in[2];   // bool[B][N]
    const float*         gamma = (const float*)d_in[3];
    const float*         beta  = (const float*)d_in[4];
    const float*         Wqkv  = (const float*)d_in[5];
    const float*         Wout  = (const float*)d_in[6];
    const float*         bout  = (const float*)d_in[7];
    float*               out   = (float*)d_out;

    // workspace layout (bytes), ~176 MB total
    char* ws = (char*)d_ws;
    _Float16* wqkv_h = (_Float16*)(ws);                 // 3*1024*1024*2 = 6291456
    _Float16* wout_h = (_Float16*)(ws + 6291456);       // 1024*1024*2   = 2097152
    _Float16* xn_h   = (_Float16*)(ws + 8388608);       // 8192*1024*2   = 16777216
    _Float16* qkv_h  = (_Float16*)(ws + 25165824);      // 8192*3072*2   = 50331648
    _Float16* q_h    = (_Float16*)(ws + 75497472);      // 64*2048*64*2  = 16777216
    _Float16* k_h    = (_Float16*)(ws + 92274688);      // 16777216
    _Float16* vt_h   = (_Float16*)(ws + 109051904);     // 16777216
    float*    o_f    = (float*)(ws + 125829120);        // 8192*1024*4   = 33554432
    _Float16* obf_h  = (_Float16*)(ws + 159383552);     // 16777216

    // 0) weights -> f16
    cvt_f32_f16_kernel<<<(3 * DMODEL * DINNER + 255) / 256, 256, 0, stream>>>(
        Wqkv, wqkv_h, 3 * DMODEL * DINNER);
    cvt_f32_f16_kernel<<<(DINNER * DMODEL + 255) / 256, 256, 0, stream>>>(
        Wout, wout_h, DINNER * DMODEL);

    // 1) LayerNorm
    ln_kernel<<<BATCH * SEQ, 256, 0, stream>>>(x, gamma, beta, xn_h);

    // 2) QKV GEMM: [8192,1024] @ [1024,3072] -> f16
    gemm_f16_kernel<<<dim3(3 * DINNER / 128, BATCH * SEQ / 128), 256, 0, stream>>>(
        xn_h, wqkv_h, qkv_h, nullptr, nullptr, BATCH * SEQ, DMODEL, 3 * DINNER);

    // 3) rotary q,k,v (+ v transpose)
    rotary_kernel<<<(BATCH * SEQ * NHEAD * (DHEAD / 2)) / 256, 256, 0, stream>>>(
        qkv_h, freqs, q_h, k_h, vt_h);

    // 4) flash attention
    attn_kernel<<<BATCH * NHEAD * (SEQ / 64), 128, 0, stream>>>(
        q_h, k_h, vt_h, mask, o_f);

    // 5) inverse rotary
    unrotary_kernel<<<(BATCH * SEQ * NHEAD * (DHEAD / 2)) / 256, 256, 0, stream>>>(
        o_f, freqs, obf_h);

    // 6) output GEMM: [8192,1024] @ [1024,1024] + bias -> f32
    gemm_f16_kernel<<<dim3(DMODEL / 128, BATCH * SEQ / 128), 256, 0, stream>>>(
        obf_h, wout_h, nullptr, out, bout, BATCH * SEQ, DINNER, DMODEL);
}
